// DLK_35218731827409
// MI455X (gfx1250) — matrix operand order repaired and checked by reading, not compiled, not run on hardware
//
#include <hip/hip_runtime.h>
#include <math.h>

typedef __attribute__((ext_vector_type(16))) _Float16 v16h;
typedef __attribute__((ext_vector_type(8)))  float    v8f;
typedef unsigned int uint32;

#define S   64
#define SS  4096        // 64*64
#define SV  262144      // 64^3
#define CH  64
#define NE  33554432ull // 2*64*64^3

// ---------------------------------------------------------------------------
// Pre-swizzle Toeplitz B operands into the contiguous-k B order this file uses:
//   B element owned by lane L, vector slot e:  K = e + 16*(L>>4), N = L&15
// Stored contiguously per lane so each wave loads its v16h with 2x b128.
// conv1 (5 taps, stride 1):  T[k,n] = w[k-n]            , 0 <= k-n < 5
// conv2 (7 taps, dil 3), win0 @col0 : T[k,n] = w[(k-n)/3], k-n in {0,3,6,9,12}
//                        win1 @col14: T[k,n] = w[5] if k-n==1, w[6] if k-n==4
// ---------------------------------------------------------------------------
__global__ __launch_bounds__(256) void dlk_b1sw(const float* __restrict__ w1,
                                                _Float16* __restrict__ Bsw) {
  int i = blockIdx.x * 256 + threadIdx.x;       // 64*25*512 elements
  int e    = i & 15;
  int lane = (i >> 4) & 31;
  int p    = (i >> 9) % 25;
  int c    = (i >> 9) / 25;
  int n = lane & 15, g = lane >> 4;
  int k = e + (g << 4);
  int t = k - n;
  float v = (t >= 0 && t < 5) ? w1[c * 125 + p * 5 + t] : 0.0f;
  Bsw[i] = (_Float16)v;
}

__global__ __launch_bounds__(256) void dlk_b2sw(const float* __restrict__ w2,
                                                _Float16* __restrict__ Bsw) {
  int i = blockIdx.x * 256 + threadIdx.x;       // 64*49*2*512 elements
  int e    = i & 15;
  int lane = (i >> 4) & 31;
  int win  = (i >> 9) & 1;
  int p    = (i >> 10) % 49;
  int c    = (i >> 10) / 49;
  int n = lane & 15, g = lane >> 4;
  int k = e + (g << 4);
  int t = k - n;
  float v = 0.0f;
  if (win == 0) {
    if (t >= 0 && t <= 12 && (t % 3) == 0) v = w2[c * 343 + p * 7 + t / 3];
  } else {
    if (t == 1)      v = w2[c * 343 + p * 7 + 5];
    else if (t == 4) v = w2[c * 343 + p * 7 + 6];
  }
  Bsw[i] = (_Float16)v;
}

// ---------------------------------------------------------------------------
// conv1: depthwise 5x5x5, pad 2.  8 waves/block share an LDS input tile
// (depths d0-2..d0+9, rows h0-2..h0+17, cols w0-2.., f16, zero-padded).
// Wave w handles depth d0+w: 25 tap-planes -> 25 x v_wmma_f32_16x16x32_f16.
// ---------------------------------------------------------------------------
__global__ __launch_bounds__(256) void dlk_conv5(const float* __restrict__ x,
    const _Float16* __restrict__ Bsw, const float* __restrict__ b1,
    float* __restrict__ att1)
{
  __shared__ _Float16 lds[12 * 20 * 32];        // 15360 B
  const int bid = blockIdx.x;
  const int w0 = (bid & 3) << 4;
  const int h0 = ((bid >> 2) & 3) << 4;
  const int d0 = ((bid >> 4) & 7) << 3;
  const int c  = (bid >> 7) & 63;
  const int b  = bid >> 13;
  const int tid = threadIdx.x;

  for (int i = tid; i < 12 * 20 * 32; i += 256) lds[i] = (_Float16)0.0f;
  __syncthreads();
  const float* xc = x + (size_t)(b * CH + c) * SV;
  for (int i = tid; i < 12 * 20 * 20; i += 256) {
    int zz = i / 400;
    int r  = i - zz * 400;
    int rr = r / 20;
    int cc = r - rr * 20;
    int dd = d0 + zz - 2, hh = h0 + rr - 2, ww = w0 + cc - 2;
    float v = 0.0f;
    if (dd >= 0 && dd < S && hh >= 0 && hh < S && ww >= 0 && ww < S)
      v = xc[dd * SS + hh * S + ww];
    lds[zz * 640 + rr * 32 + cc] = (_Float16)v;
  }
  __syncthreads();

  const int wave = tid >> 5;
  const int lane = tid & 31;
  const int m = lane & 15;                      // output h-row (M) / N col
  const int g = lane >> 4;

  v8f acc = {};
  const _Float16* Bc = Bsw + (size_t)c * (25 * 512) + lane * 16;
  for (int kd = 0; kd < 5; ++kd) {
    for (int kh = 0; kh < 5; ++kh) {
      const uint32* row32 = (const uint32*)&lds[(wave + kd) * 640 + (m + kh) * 32];
      union { uint32 u[8]; v16h h; } au;
      #pragma unroll
      for (int e2 = 0; e2 < 8; ++e2) {          // A layout: K pairs per VGPR
        int k = (e2 << 1) + (g << 4);
        au.u[e2] = row32[k >> 1];
      }
      v16h bm = *(const v16h*)(Bc + (kd * 5 + kh) * 512);
      acc = __builtin_amdgcn_wmma_f32_16x16x32_f16(false, au.h, false, bm,
                                                   (short)0, acc, false, false);
    }
  }
  const float bias = b1[c];
  float* outp = att1 + (size_t)(b * CH + c) * SV + (d0 + wave) * SS;
  #pragma unroll
  for (int r = 0; r < 8; ++r) {                 // D layout: M = r + 8g, N = m
    int hh = h0 + r + (g << 3);
    outp[hh * S + w0 + m] = acc[r] + bias;
  }
}

// ---------------------------------------------------------------------------
// conv2: depthwise 7x7x7 dilation 3, pad 9.  LDS tile: depths d0-9..d0+16,
// rows h0-9..h0+24, col stride 48 (window1 reads up to col 45), f16, zeroed.
// Per tap-plane: two WMMAs (windows @col0 and @col14).
// ---------------------------------------------------------------------------
__global__ __launch_bounds__(256) void dlk_conv7d3(const float* __restrict__ att1,
    const _Float16* __restrict__ Bsw, const float* __restrict__ b2,
    float* __restrict__ att2)
{
  __shared__ _Float16 lds[26 * 34 * 48];        // 84864 B (<320KB/WGP)
  const int bid = blockIdx.x;
  const int w0 = (bid & 3) << 4;
  const int h0 = ((bid >> 2) & 3) << 4;
  const int d0 = ((bid >> 4) & 7) << 3;
  const int c  = (bid >> 7) & 63;
  const int b  = bid >> 13;
  const int tid = threadIdx.x;

  for (int i = tid; i < 26 * 34 * 48; i += 256) lds[i] = (_Float16)0.0f;
  __syncthreads();
  const float* ac = att1 + (size_t)(b * CH + c) * SV;
  for (int i = tid; i < 26 * 34 * 34; i += 256) {
    int zz = i / 1156;
    int r  = i - zz * 1156;
    int rr = r / 34;
    int cc = r - rr * 34;
    int dd = d0 + zz - 9, hh = h0 + rr - 9, ww = w0 + cc - 9;
    float v = 0.0f;
    if (dd >= 0 && dd < S && hh >= 0 && hh < S && ww >= 0 && ww < S)
      v = ac[dd * SS + hh * S + ww];
    lds[zz * 1632 + rr * 48 + cc] = (_Float16)v;
  }
  __syncthreads();

  const int wave = tid >> 5;
  const int lane = tid & 31;
  const int m = lane & 15;
  const int g = lane >> 4;

  v8f acc = {};
  const _Float16* Bc = Bsw + (size_t)c * (49 * 2 * 512) + lane * 16;
  for (int kd = 0; kd < 7; ++kd) {
    const int zz = wave + 3 * kd;               // depth d0+wave + 3*(kd-3)
    for (int kh = 0; kh < 7; ++kh) {
      const _Float16* rowh = &lds[zz * 1632 + (m + 3 * kh) * 48];
      const uint32* rowa = (const uint32*)rowh;        // window 0 @ col 0
      const uint32* rowb = (const uint32*)(rowh + 14); // window 1 @ col 14
      const int p = kd * 7 + kh;
      union { uint32 u[8]; v16h h; } a0, a1;
      #pragma unroll
      for (int e2 = 0; e2 < 8; ++e2) {
        int k = (e2 << 1) + (g << 4);
        a0.u[e2] = rowa[k >> 1];
        a1.u[e2] = rowb[k >> 1];
      }
      v16h bm0 = *(const v16h*)(Bc + (p * 2 + 0) * 512);
      v16h bm1 = *(const v16h*)(Bc + (p * 2 + 1) * 512);
      acc = __builtin_amdgcn_wmma_f32_16x16x32_f16(false, a0.h, false, bm0,
                                                   (short)0, acc, false, false);
      acc = __builtin_amdgcn_wmma_f32_16x16x32_f16(false, a1.h, false, bm1,
                                                   (short)0, acc, false, false);
    }
  }
  const float bias = b2[c];
  float* outp = att2 + (size_t)(b * CH + c) * SV + (d0 + wave) * SS;
  #pragma unroll
  for (int r = 0; r < 8; ++r) {
    int hh = h0 + r + (g << 3);
    outp[hh * S + w0 + m] = acc[r] + bias;
  }
}

// ---------------------------------------------------------------------------
// Channel mean/max over [att1;att2] (128 channels) -> pooled [B,2,D,H,W]
// ---------------------------------------------------------------------------
__global__ __launch_bounds__(256) void dlk_pool(const float* __restrict__ att1,
    const float* __restrict__ att2, float* __restrict__ pooled)
{
  int idx = blockIdx.x * 256 + threadIdx.x;
  int stride = gridDim.x * 256;
  for (; idx < 2 * SV; idx += stride) {
    int b = idx >> 18;
    int s = idx & (SV - 1);
    const float* p1 = att1 + (size_t)b * CH * SV + s;
    const float* p2 = att2 + (size_t)b * CH * SV + s;
    float sum = 0.0f, mx = -INFINITY;
    for (int c = 0; c < CH; ++c) {
      float a = p1[(size_t)c * SV];
      float e = p2[(size_t)c * SV];
      sum += a + e;
      mx = fmaxf(mx, fmaxf(a, e));
    }
    pooled[(size_t)(b * 2) * SV + s]     = sum * (1.0f / 128.0f);
    pooled[(size_t)(b * 2 + 1) * SV + s] = mx;
  }
}

// ---------------------------------------------------------------------------
// Gate conv (2->2, 7^3, pad 3) + sigmoid + combine: out = a1*g0 + a2*g1 + x
// ---------------------------------------------------------------------------
__global__ __launch_bounds__(256) void dlk_gate(const float* __restrict__ x,
    const float* __restrict__ att1, const float* __restrict__ att2,
    const float* __restrict__ pooled, const float* __restrict__ wg,
    const float* __restrict__ bg, float* __restrict__ out)
{
  __shared__ float tile[2 * 7 * 22 * 22];       // 27104 B
  const int bid = blockIdx.x;
  const int w0 = (bid & 3) << 4;
  const int h0 = ((bid >> 2) & 3) << 4;
  const int d  = (bid >> 4) & 63;
  const int b  = bid >> 10;
  const int tid = threadIdx.x;

  for (int i = tid; i < 2 * 7 * 22 * 22; i += 256) {
    int ch = i / 3388;
    int r  = i - ch * 3388;
    int zz = r / 484; r -= zz * 484;
    int rr = r / 22;
    int cc = r - rr * 22;
    int dd = d + zz - 3, hh = h0 + rr - 3, ww = w0 + cc - 3;
    float v = 0.0f;
    if (dd >= 0 && dd < S && hh >= 0 && hh < S && ww >= 0 && ww < S)
      v = pooled[(size_t)(b * 2 + ch) * SV + dd * SS + hh * S + ww];
    tile[i] = v;
  }
  __syncthreads();

  const int wl = tid & 15;
  const int hl = tid >> 4;
  float g0 = bg[0], g1 = bg[1];
  for (int ci = 0; ci < 2; ++ci)
    for (int kd = 0; kd < 7; ++kd)
      for (int kh = 0; kh < 7; ++kh) {
        const float* trow = &tile[ci * 3388 + kd * 484 + (hl + kh) * 22 + wl];
        const float* wrow = &wg[ci * 343 + kd * 49 + kh * 7];
        #pragma unroll
        for (int kw = 0; kw < 7; ++kw) {
          float v = trow[kw];
          g0 = fmaf(wrow[kw], v, g0);
          g1 = fmaf(wrow[686 + kw], v, g1);
        }
      }
  g0 = 1.0f / (1.0f + __expf(-g0));
  g1 = 1.0f / (1.0f + __expf(-g1));

  const size_t sp = (size_t)d * SS + (size_t)(h0 + hl) * S + (w0 + wl);
  for (int c = 0; c < CH; ++c) {
    size_t o = (size_t)(b * CH + c) * SV + sp;
    out[o] = fmaf(att1[o], g0, fmaf(att2[o], g1, x[o]));
  }
}

// ---------------------------------------------------------------------------
// Scratch layout (bytes): att1 fp32 (134MB) | att2 fp32 (134MB) |
//   pooled fp32 (4MB) | B1sw f16 (1.6MB) | B2sw f16 (6.4MB)  ~= 281 MB
// ---------------------------------------------------------------------------
extern "C" void kernel_launch(void* const* d_in, const int* in_sizes, int n_in,
                              void* d_out, int out_size, void* d_ws, size_t ws_size,
                              hipStream_t stream) {
  (void)in_sizes; (void)n_in; (void)out_size; (void)ws_size;
  const float* x  = (const float*)d_in[0];
  const float* w1 = (const float*)d_in[1];
  const float* b1 = (const float*)d_in[2];
  const float* w2 = (const float*)d_in[3];
  const float* b2 = (const float*)d_in[4];
  const float* wg = (const float*)d_in[5];
  const float* bg = (const float*)d_in[6];
  float* out = (float*)d_out;

  float* att1   = (float*)d_ws;
  float* att2   = att1 + NE;
  float* pooled = att2 + NE;
  _Float16* B1sw = (_Float16*)(pooled + 4ull * SV);
  _Float16* B2sw = B1sw + 64 * 25 * 512;

  dlk_b1sw<<<(64 * 25 * 512) / 256, 256, 0, stream>>>(w1, B1sw);
  dlk_b2sw<<<(64 * 49 * 2 * 512) / 256, 256, 0, stream>>>(w2, B2sw);
  dlk_conv5<<<16384, 256, 0, stream>>>(x, B1sw, b1, att1);
  dlk_conv7d3<<<16384, 256, 0, stream>>>(att1, B2sw, b2, att2);
  dlk_pool<<<2048, 256, 0, stream>>>(att1, att2, pooled);
  dlk_gate<<<2048, 256, 0, stream>>>(x, att1, att2, pooled, wg, bg, out);
}
